// SingleHeadSelfAttention_34952443855229
// MI455X (gfx1250) — compile-verified, matrix operand order checked by reading, not run on hardware
//
#include <hip/hip_runtime.h>
#include <hip/hip_bf16.h>
#include <stdint.h>

#define TSEQ 2048
#define DDIM 512
#define NEG_FILL (-2147483648.0f)
#define SCALE 0.04419417382415922f  // 1/sqrt(512)

#if __has_builtin(__builtin_amdgcn_tensor_load_to_lds)
#define HAVE_TDM 1
#else
#define HAVE_TDM 0
#endif
#if __has_include(<hip/amd_detail/amd_gfx1250_TDM.h>)
#define TDM_6ARG 1
#else
#define TDM_6ARG 0
#endif

typedef __attribute__((ext_vector_type(16))) __bf16 v16bf;
typedef __attribute__((ext_vector_type(8)))  float v8f;
typedef __attribute__((ext_vector_type(4)))  unsigned int u32x4;
typedef __attribute__((ext_vector_type(8)))  int i32x8;
typedef __attribute__((ext_vector_type(4)))  int i32x4;

union FragAB {
    v16bf v;
    uint4 u[2];
    unsigned int w[8];
};
union FragC {
    v8f v;
    float f[8];
};

__device__ inline unsigned short f2bf(float x) {
    unsigned int u = __float_as_uint(x);
    u = (u + 0x7FFFu + ((u >> 16) & 1u)) >> 16;
    return (unsigned short)u;
}

__device__ inline v8f wmma_bf16(const FragAB& a, const FragAB& b, v8f c) {
    return __builtin_amdgcn_wmma_f32_16x16x32_bf16(
        false, a.v, false, b.v, (short)0, c, false, false);
}

// 16(row)x32(k) bf16 fragment. Lane L<16 owns row L with k={ko..ko+7, ko+16..ko+23},
// ko=(L>>4)*8, per CDNA5 16-bit A-matrix layout. B operand loaded identically from Bt rows.
__device__ inline FragAB load_frag(const unsigned short* base, int ld, int row0, int lane) {
    int r = row0 + (lane & 15);
    int ko = (lane >> 4) * 8;
    const unsigned short* p = base + (size_t)r * ld + ko;
    FragAB f;
    f.u[0] = *(const uint4*)(p);
    f.u[1] = *(const uint4*)(p + 16);
    return f;
}

__device__ inline FragAB frag_from_lds(const unsigned short* p) {
    FragAB f;
    f.u[0] = *(const uint4*)(p);
    f.u[1] = *(const uint4*)(p + 16);
    return f;
}

__device__ inline v8f zero8() {
    v8f z = {0.f, 0.f, 0.f, 0.f, 0.f, 0.f, 0.f, 0.f};
    return z;
}

template <int N>
__device__ __forceinline__ void wait_tensorcnt() {
#if __has_builtin(__builtin_amdgcn_s_wait_tensorcnt)
    __builtin_amdgcn_s_wait_tensorcnt((short)N);
#else
    asm volatile("s_wait_tensorcnt %0" ::"i"(N) : "memory");
#endif
}

#if HAVE_TDM
// Issue one TDM 2D tile load: rows x tile_k bf16 elements, row stride = ld elements,
// packed contiguously into LDS (rows*tile_k*2 bytes). D# per CDNA5 ISA ch.8.
__device__ __forceinline__ void tdm_load_tile_2d(const unsigned short* g,
                                                 unsigned int lds_bytes, int rows,
                                                 int ld, int tile_k) {
    unsigned long long ga = (unsigned long long)(size_t)g;
    u32x4 g0;
    g0[0] = 1u;                                            // count=1, user mode
    g0[1] = lds_bytes;                                     // lds_addr
    g0[2] = (unsigned int)ga;                              // global_addr[31:0]
    g0[3] = (unsigned int)((ga >> 32) & 0x01FFFFFFu) | 0x80000000u;  // addr[56:32], type=2
    i32x8 g1;
    g1[0] = 0x00010000;                                    // mask=0, data_size=1 (2 bytes)
    g1[1] = (int)(((unsigned)ld & 0xFFFFu) << 16);         // tensor_dim0[15:0]
    g1[2] = (int)((((unsigned)ld >> 16) & 0xFFFFu) | (((unsigned)rows & 0xFFFFu) << 16));
    g1[3] = (int)((((unsigned)rows >> 16) & 0xFFFFu) | (((unsigned)tile_k & 0xFFFFu) << 16));
    g1[4] = rows;                                          // tile_dim1 = rows
    g1[5] = ld;                                            // tensor_dim0_stride[31:0]
    g1[6] = 0;
    g1[7] = 0;
    i32x4 z4 = {0, 0, 0, 0};
#if TDM_6ARG
    i32x8 z8 = {0, 0, 0, 0, 0, 0, 0, 0};
    __builtin_amdgcn_tensor_load_to_lds(g0, g1, z4, z4, z8, 0);
#else
    __builtin_amdgcn_tensor_load_to_lds(g0, g1, z4, z4, 0);
#endif
}
__device__ __forceinline__ unsigned int lds_off(const void* p) {
    return (unsigned int)(size_t)p;  // generic LDS address: low 32 bits = LDS byte offset
}
#endif

// ---------------- conversion kernels ----------------

__global__ __launch_bounds__(256) void cvt_bf16_kernel(const float* __restrict__ in,
                                                       unsigned short* __restrict__ out, int n) {
    int i = blockIdx.x * 256 + threadIdx.x;
    if (i < n) out[i] = f2bf(in[i]);
}

// Wt[n*512 + k] = bf16(W[k*512 + n])
__global__ __launch_bounds__(256) void transpose_cvt_kernel(const float* __restrict__ W,
                                                            unsigned short* __restrict__ Wt) {
    int idx = blockIdx.x * 256 + threadIdx.x;  // 512*512 total
    int k = idx >> 9;
    int n = idx & 511;
    Wt[(size_t)n * 512 + k] = f2bf(W[idx]);
}

// Sinusoidal PE rows 0..2047 (slice [MAX_LEN-T : MAX_LEN] of the 4096 table == rows 0..2047)
__global__ __launch_bounds__(256) void pe_kernel(unsigned short* __restrict__ PEb) {
    int idx = blockIdx.x * 256 + threadIdx.x;  // 2048*512
    int r = idx >> 9;
    int c = idx & 511;
    int m = (c < 256) ? c : c - 256;
    const float sf = -9.210340371976184f / 255.0f;  // -ln(10000)/(256-1)
    float dv = __expf((float)m * sf);
    float ang = (float)r * dv;
    float v = (c < 256) ? __sinf(ang) : __cosf(ang);
    PEb[idx] = f2bf(v);
}

// ---------------- projection GEMM: C(MxN) = A(MxK) * Bt(NxK)^T, N=K=512 ----------------
// TDM path: A/B k-slices staged in LDS via tensor_load_to_lds, double-buffered,
// synchronized with s_wait_tensorcnt + workgroup barriers.
// MODE 0: out0[m,n]; MODE 1: dual-bias -> out0/out1; MODE 2: transposed store out0[n*M+m].
template <int MODE>
__global__ __launch_bounds__(256) void gemm_proj_kernel(const unsigned short* __restrict__ A,
                                                        const unsigned short* __restrict__ Bt,
                                                        unsigned short* __restrict__ out0,
                                                        unsigned short* __restrict__ out1,
                                                        const float* __restrict__ bias0,
                                                        const float* __restrict__ bias1, int M) {
    int wave = threadIdx.x >> 5;
    int lane = threadIdx.x & 31;
    int m_blk = blockIdx.x * 128;
    int n_blk = blockIdx.y * 64;
    int m_wl = (wave >> 1) * 32;  // local within 128
    int n_wl = (wave & 1) * 32;   // local within 64

    FragC acc[2][2];
    acc[0][0].v = zero8(); acc[0][1].v = zero8();
    acc[1][0].v = zero8(); acc[1][1].v = zero8();

#if HAVE_TDM
    __shared__ unsigned short ldsA[2][128 * 32];  // 2 x 8 KiB
    __shared__ unsigned short ldsB[2][64 * 32];   // 2 x 4 KiB
    const unsigned short* Ab = A + (size_t)m_blk * 512;
    const unsigned short* Bb = Bt + (size_t)n_blk * 512;
    if (wave == 0) {
        tdm_load_tile_2d(Ab, lds_off(&ldsA[0][0]), 128, 512, 32);
        tdm_load_tile_2d(Bb, lds_off(&ldsB[0][0]), 64, 512, 32);
    }
    for (int ks = 0; ks < 16; ks++) {
        int buf = ks & 1;
        if (wave == 0) {
            if (ks + 1 < 16) {
                tdm_load_tile_2d(Ab + (ks + 1) * 32, lds_off(&ldsA[buf ^ 1][0]), 128, 512, 32);
                tdm_load_tile_2d(Bb + (ks + 1) * 32, lds_off(&ldsB[buf ^ 1][0]), 64, 512, 32);
                wait_tensorcnt<2>();  // this buffer's two loads done; next two in flight
            } else {
                wait_tensorcnt<0>();
            }
        }
        __syncthreads();
        int r = lane & 15;
        int ko = (lane >> 4) * 8;
        FragAB a0 = frag_from_lds(&ldsA[buf][(m_wl + r) * 32 + ko]);
        FragAB a1 = frag_from_lds(&ldsA[buf][(m_wl + 16 + r) * 32 + ko]);
        FragAB b0 = frag_from_lds(&ldsB[buf][(n_wl + r) * 32 + ko]);
        FragAB b1 = frag_from_lds(&ldsB[buf][(n_wl + 16 + r) * 32 + ko]);
        acc[0][0].v = wmma_bf16(a0, b0, acc[0][0].v);
        acc[0][1].v = wmma_bf16(a0, b1, acc[0][1].v);
        acc[1][0].v = wmma_bf16(a1, b0, acc[1][0].v);
        acc[1][1].v = wmma_bf16(a1, b1, acc[1][1].v);
        __syncthreads();
    }
#else
#pragma unroll 4
    for (int kk = 0; kk < 512; kk += 32) {
        FragAB a0 = load_frag(A + kk, 512, m_blk + m_wl, lane);
        FragAB a1 = load_frag(A + kk, 512, m_blk + m_wl + 16, lane);
        FragAB b0 = load_frag(Bt + kk, 512, n_blk + n_wl, lane);
        FragAB b1 = load_frag(Bt + kk, 512, n_blk + n_wl + 16, lane);
        acc[0][0].v = wmma_bf16(a0, b0, acc[0][0].v);
        acc[0][1].v = wmma_bf16(a0, b1, acc[0][1].v);
        acc[1][0].v = wmma_bf16(a1, b0, acc[1][0].v);
        acc[1][1].v = wmma_bf16(a1, b1, acc[1][1].v);
    }
#endif

    int col = lane & 15;
    int rb = (lane >> 4) * 8;
#pragma unroll
    for (int i = 0; i < 2; i++) {
#pragma unroll
        for (int j = 0; j < 2; j++) {
            int n = n_blk + n_wl + j * 16 + col;
#pragma unroll
            for (int v = 0; v < 8; v++) {
                int m = m_blk + m_wl + i * 16 + rb + v;
                float x = acc[i][j].f[v];
                if (MODE == 1) {
                    out0[(size_t)m * 512 + n] = f2bf(x + bias0[n]);
                    out1[(size_t)m * 512 + n] = f2bf(x + bias1[n]);
                } else if (MODE == 2) {
                    out0[(size_t)n * M + m] = f2bf(x);
                } else {
                    out0[(size_t)m * 512 + n] = f2bf(x);
                }
            }
        }
    }
}

// ---------------- fused attention: logits -> softmax -> P*V, one row-block per workgroup ----
// 16 query rows per workgroup; the whole 2048-key fp32 logit row lives in LDS (128 KB of the
// 320 KB WGP LDS). Q fragments register-resident; K/RK/V fragment loads software-pipelined
// (ping-pong registers) so each global_load_b128 pair overlaps the previous WMMA.
__global__ __launch_bounds__(256) void attn_kernel(const unsigned short* __restrict__ Qc,
                                                   const unsigned short* __restrict__ Qr,
                                                   const unsigned short* __restrict__ Kb,
                                                   const unsigned short* __restrict__ Vt,
                                                   const unsigned short* __restrict__ RK,
                                                   float* __restrict__ out) {
    __shared__ float S[16 * TSEQ];  // 131072 bytes
    __shared__ float rowmax[16];
    __shared__ float rowsum[16];

    int b = blockIdx.y;
    int i0 = blockIdx.x * 16;
    int wave = threadIdx.x >> 5;
    int lane = threadIdx.x & 31;

    const unsigned short* Qcb = Qc + (size_t)(b * TSEQ + i0) * DDIM;
    const unsigned short* Qrb = Qr + (size_t)(b * TSEQ + i0) * DDIM;
    const unsigned short* Kbb = Kb + (size_t)b * TSEQ * DDIM;
    const unsigned short* Vtb = Vt + (size_t)b * TSEQ;  // Vt[d * 16384 + b*2048 + t]

    int nt = i0 / 16 + 1;  // causal key tiles
    int Lb = nt * 16;      // valid key extent (== i0 + 16)
    int col = lane & 15;
    int rb = (lane >> 4) * 8;

    // -------- Phase A1: content logits. Q fragments hoisted to registers.
    FragAB qf[16];
#pragma unroll
    for (int ks = 0; ks < 16; ks++) qf[ks] = load_frag(Qcb + ks * 32, DDIM, 0, lane);
    for (int t = wave; t < nt; t += 8) {
        if (t + 8 < nt)  // prefetch this wave's next K tile (global_prefetch_b8)
            __builtin_prefetch(Kbb + (size_t)(t + 8) * 16 * DDIM + (size_t)(lane & 15) * DDIM, 0, 0);
        const unsigned short* Kt = Kbb + (size_t)t * 16 * DDIM;
        FragC acc; acc.v = zero8();
        FragAB kcur = load_frag(Kt, DDIM, 0, lane);
#pragma unroll
        for (int ks = 0; ks < 16; ks++) {
            FragAB knext = kcur;
            if (ks + 1 < 16) knext = load_frag(Kt + (ks + 1) * 32, DDIM, 0, lane);
            acc.v = wmma_bf16(qf[ks], kcur, acc.v);
            kcur = knext;
        }
        int j0 = t * 16;
        bool diag = (j0 == i0);
#pragma unroll
        for (int v = 0; v < 8; v++) {
            int r = rb + v;
            float val = acc.f[v] * SCALE;
            if (diag && col > r) val = NEG_FILL;
            S[r * TSEQ + j0 + col] = val;
        }
    }
    __syncthreads();

    // -------- Phase A2: relative logits. G[i,u] = Qr_i . RK[u]; scatter to j = u-(T-1)+i
    // (bijective (i,u)->(i,j): no write collisions, so plain LDS RMW is safe).
#pragma unroll
    for (int ks = 0; ks < 16; ks++) qf[ks] = load_frag(Qrb + ks * 32, DDIM, 0, lane);
    int ubase0 = TSEQ - 16 - i0;
    for (int t = wave; t < nt; t += 8) {
        int u0 = ubase0 + t * 16;
        const unsigned short* Rt = RK + (size_t)u0 * DDIM;
        FragC acc; acc.v = zero8();
        FragAB rcur = load_frag(Rt, DDIM, 0, lane);
#pragma unroll
        for (int ks = 0; ks < 16; ks++) {
            FragAB rnext = rcur;
            if (ks + 1 < 16) rnext = load_frag(Rt + (ks + 1) * 32, DDIM, 0, lane);
            acc.v = wmma_bf16(qf[ks], rcur, acc.v);
            rcur = rnext;
        }
#pragma unroll
        for (int v = 0; v < 8; v++) {
            int r = rb + v;
            int j = (u0 + col) - (TSEQ - 1) + (i0 + r);  // always <= i
            if (j >= 0) S[r * TSEQ + j] += acc.f[v] * SCALE;
        }
    }
    __syncthreads();

    // -------- Phase B: per-row max and sum(exp). Wave w handles rows 2w, 2w+1.
    {
        int r = wave * 2 + (lane >> 4);
        int c0 = lane & 15;
        float m = -3.4e38f;
        for (int j = c0; j < Lb; j += 16) m = fmaxf(m, S[r * TSEQ + j]);
        m = fmaxf(m, __shfl_xor(m, 8, 32));
        m = fmaxf(m, __shfl_xor(m, 4, 32));
        m = fmaxf(m, __shfl_xor(m, 2, 32));
        m = fmaxf(m, __shfl_xor(m, 1, 32));
        float s = 0.f;
        for (int j = c0; j < Lb; j += 16) s += __expf(S[r * TSEQ + j] - m);
        s += __shfl_xor(s, 8, 32);
        s += __shfl_xor(s, 4, 32);
        s += __shfl_xor(s, 2, 32);
        s += __shfl_xor(s, 1, 32);
        if (c0 == 0) { rowmax[r] = m; rowsum[r] = s; }
    }
    __syncthreads();

    // -------- Phase C: O = P * V. Wave w owns d-tiles [4w, 4w+4). P built from LDS on the fly,
    // normalization by 1/rowsum deferred to the epilogue (flash-attention style).
    FragC o[4];
    o[0].v = zero8(); o[1].v = zero8(); o[2].v = zero8(); o[3].v = zero8();
    int nk = (Lb + 31) / 32;
    for (int kss = 0; kss < nk; kss++) {
        int kk = kss * 32;
        FragAB a;
        int r = lane & 15;
        int ko = (lane >> 4) * 8;
        float rm = rowmax[r];
#pragma unroll
        for (int q = 0; q < 8; q++) {
            int ka = (q < 4) ? (ko + 2 * q) : (ko + 16 + 2 * (q - 4));
            int k0 = kk + ka;
            float p0 = (k0 < Lb) ? __expf(S[r * TSEQ + k0] - rm) : 0.f;
            float p1 = (k0 + 1 < Lb) ? __expf(S[r * TSEQ + k0 + 1] - rm) : 0.f;
            a.w[q] = (unsigned int)f2bf(p0) | ((unsigned int)f2bf(p1) << 16);
        }
        const unsigned short* Vk = Vtb + kk;
        FragAB vcur = load_frag(Vk, 16384, wave * 64, lane);
#pragma unroll
        for (int c = 0; c < 4; c++) {
            FragAB vnext = vcur;
            if (c + 1 < 4) vnext = load_frag(Vk, 16384, (wave * 4 + c + 1) * 16, lane);
            o[c].v = wmma_bf16(a, vcur, o[c].v);
            vcur = vnext;
        }
    }
#pragma unroll
    for (int c = 0; c < 4; c++) {
        int d = (wave * 4 + c) * 16 + col;
#pragma unroll
        for (int v = 0; v < 8; v++) {
            int r = rb + v;
            out[((size_t)(b * TSEQ) + i0 + r) * DDIM + d] = o[c].f[v] / rowsum[r];
        }
    }
}

// ---------------- host launch ----------------

extern "C" void kernel_launch(void* const* d_in, const int* in_sizes, int n_in,
                              void* d_out, int out_size, void* d_ws, size_t ws_size,
                              hipStream_t stream) {
    (void)in_sizes; (void)n_in; (void)out_size; (void)ws_size;
    const float* X    = (const float*)d_in[0];
    const float* Wq   = (const float*)d_in[1];
    const float* Wk   = (const float*)d_in[2];
    const float* Wv   = (const float*)d_in[3];
    const float* Wrel = (const float*)d_in[4];
    const float* cb   = (const float*)d_in[5];
    const float* rbv  = (const float*)d_in[6];
    float* out = (float*)d_out;

    char* ws = (char*)d_ws;
    // bf16 workspace layout (bytes)
    unsigned short* Xb    = (unsigned short*)(ws + 0);          // 16384*512*2 = 16 MiB
    unsigned short* Qc    = (unsigned short*)(ws + 16777216);
    unsigned short* Qr    = (unsigned short*)(ws + 33554432);
    unsigned short* Kb    = (unsigned short*)(ws + 50331648);
    unsigned short* Vt    = (unsigned short*)(ws + 67108864);
    unsigned short* WqT   = (unsigned short*)(ws + 83886080);   // 512*512*2
    unsigned short* WkT   = (unsigned short*)(ws + 84410368);
    unsigned short* WvT   = (unsigned short*)(ws + 84934656);
    unsigned short* WrelT = (unsigned short*)(ws + 85458944);
    unsigned short* PEb   = (unsigned short*)(ws + 85983232);   // 2048*512*2
    unsigned short* RKb   = (unsigned short*)(ws + 88080384);   // 2048*512*2

    const int NX = 16384 * 512;
    cvt_bf16_kernel<<<(NX + 255) / 256, 256, 0, stream>>>(X, Xb, NX);
    transpose_cvt_kernel<<<1024, 256, 0, stream>>>(Wq, WqT);
    transpose_cvt_kernel<<<1024, 256, 0, stream>>>(Wk, WkT);
    transpose_cvt_kernel<<<1024, 256, 0, stream>>>(Wv, WvT);
    transpose_cvt_kernel<<<1024, 256, 0, stream>>>(Wrel, WrelT);
    pe_kernel<<<(2048 * 512) / 256, 256, 0, stream>>>(PEb);

    // Projections: Q (dual bias -> Qc,Qr), K, V^T, relative key
    gemm_proj_kernel<1><<<dim3(128, 8), 256, 0, stream>>>(Xb, WqT, Qc, Qr, cb, rbv, 16384);
    gemm_proj_kernel<0><<<dim3(128, 8), 256, 0, stream>>>(Xb, WkT, Kb, nullptr, nullptr, nullptr, 16384);
    gemm_proj_kernel<2><<<dim3(128, 8), 256, 0, stream>>>(Xb, WvT, Vt, nullptr, nullptr, nullptr, 16384);
    gemm_proj_kernel<0><<<dim3(16, 8), 256, 0, stream>>>(PEb, WrelT, RKb, nullptr, nullptr, nullptr, 2048);

    // Fused attention: one 16-query row-block per workgroup, full key row in LDS.
    attn_kernel<<<dim3(TSEQ / 16, 8), 256, 0, stream>>>(Qc, Qr, Kb, Vt, RKb, out);
}
